// LSTMModel_70300024701466
// MI455X (gfx1250) — matrix operand order repaired and checked by reading, not compiled, not run on hardware
//
#include <hip/hip_runtime.h>

// LSTM: B=512, T=512, F=32, H=128, L=5. Fully fused single kernel.
// g = [x_t | h] @ [W_ih | W_hh]^T + b, K = 160, bf16 WMMA, weights in VGPRs.
// Ping-pong A-tile in LDS -> ONE barrier per time step.

#define BATCH 512
#define TSEQ  512
#define FIN   32
#define HID   128
#define LOUT  5
#define KTOT  (FIN + HID)   // 160
#define ATILE (16 * KTOT)   // elements per A buffer

typedef __attribute__((ext_vector_type(16))) __bf16 v16bf;
typedef __attribute__((ext_vector_type(8)))  float  v8f;

__device__ __forceinline__ unsigned short f2bf(float f) {
    union { float f; unsigned u; } x; x.f = f;
    unsigned u = x.u;
    u += 0x7FFFu + ((u >> 16) & 1u);   // round-to-nearest-even
    return (unsigned short)(u >> 16);
}

__device__ __forceinline__ float fsigmoid(float v) {
    return 1.0f / (1.0f + __expf(-v));
}
__device__ __forceinline__ float ftanh(float v) {
    return 2.0f / (1.0f + __expf(-2.0f * v)) - 1.0f;
}

__global__ __launch_bounds__(256, 1)
void lstm_fused_wmma(const float* __restrict__ x,
                     const float* __restrict__ W_ih,
                     const float* __restrict__ W_hh,
                     const float* __restrict__ b_ih,
                     const float* __restrict__ b_hh,
                     const float* __restrict__ W_fc,
                     const float* __restrict__ b_fc,
                     float* __restrict__ out)
{
    // Two A-tiles [16 rows][K=160] bf16 (cols 0..31 = x_t, 32..159 = h), ping-pong.
    __shared__ __align__(16) unsigned short sA[2 * ATILE];
    __shared__ float sH[16 * HID];     // final-step h in f32 for the head

    const int tid  = threadIdx.x;
    const int lane = tid & 31;
    const int wv   = tid >> 5;         // wave 0..7
    const int lmod = lane & 15;        // N / M index inside tile
    const int lhi  = lane >> 4;        // 0 or 1 (lane half)
    const int row0 = blockIdx.x * 16;  // global batch row base

    // ---------------- load B fragments (weights) into registers ----------------
    // Wave w owns gate tiles at n0 = g*128 + w*16 (g = i,f,g,o): all four gates
    // for the same 16 h-columns -> in-register state update, no gate LDS traffic.
    v16bf bfrag[4][5];
    float bias[4];
    #pragma unroll
    for (int g = 0; g < 4; ++g) {
        const int n = g * HID + wv * 16 + lmod;
        bias[g] = b_ih[n] + b_hh[n];
        #pragma unroll
        for (int kk = 0; kk < 5; ++kk) {
            const int koff = kk * 32 + lhi * 16;   // 16 consecutive K per lane
            union { unsigned short s[16]; v16bf v; } u;
            #pragma unroll
            for (int i = 0; i < 16; ++i) {
                const int k = koff + i;
                const float w = (k < FIN) ? W_ih[n * FIN + k]
                                          : W_hh[n * HID + (k - FIN)];
                u.s[i] = f2bf(w);
            }
            bfrag[g][kk] = u.v;
        }
    }

    // ---------------- init h = 0 in buffer 0, c = 0 in registers ----------------
    for (int i = tid; i < 16 * HID; i += 256) {
        const int m = i >> 7, n = i & (HID - 1);
        sA[m * KTOT + FIN + n] = 0;
    }
    float cc[8];
    #pragma unroll
    for (int r = 0; r < 8; ++r) cc[r] = 0.0f;

    // per-thread x feed: row xm (0..15), 2 floats at col xk
    const int xm = tid >> 4;
    const int xk = (tid & 15) * 2;
    const float* xrow = x + (size_t)(row0 + xm) * TSEQ * FIN + xk;

    const int arow = lmod * KTOT;          // A row base for this lane (M = lmod)
    const int col  = wv * 16 + lmod;       // h column owned by this thread

    // ================= recurrence over T (one barrier per step) =================
    for (int t = 0; t < TSEQ; ++t) {
        unsigned short* cur = sA + (t & 1) * ATILE;         // read buffer
        unsigned short* nxt = sA + ((t + 1) & 1) * ATILE;   // write buffer

        // stage x_t into CURRENT buffer (2 bf16 packed as one b32 store)
        const float2 xv = *(const float2*)(xrow + (size_t)t * FIN);
        const unsigned pack = (unsigned)f2bf(xv.x) | ((unsigned)f2bf(xv.y) << 16);
        *(unsigned*)&cur[xm * KTOT + xk] = pack;
        if (t + 1 < TSEQ)
            __builtin_prefetch(xrow + (size_t)(t + 1) * FIN, 0, 3);

        __syncthreads();   // x_t (this iter) + h (prev iter) visible before reads

        // accumulators start at 0 (bias folded into gate math below)
        v8f acc[4] = {};

        // K loop: 5 x (1 A-fragment shared by 4 gate WMMAs)
        #pragma unroll
        for (int kk = 0; kk < 5; ++kk) {
            const int e0 = kk * 32 + lhi * 16;
            union { uint4 q[2]; v16bf v; } ua;
            ua.q[0] = *(const uint4*)&cur[arow + e0];
            ua.q[1] = *(const uint4*)&cur[arow + e0 + 8];
            #pragma unroll
            for (int g = 0; g < 4; ++g) {
                acc[g] = __builtin_amdgcn_wmma_f32_16x16x32_bf16(
                    false, ua.v, false, bfrag[g][kk],
                    (short)0, acc[g], false, false);
            }
        }

        // gate nonlinearity + state update, in-register; h_{t+1} -> NEXT buffer
        // (different LDS buffer than this step's reads -> no WAR barrier needed)
        #pragma unroll
        for (int r = 0; r < 8; ++r) {
            const float is = fsigmoid(acc[0][r] + bias[0]);
            const float fs = fsigmoid(acc[1][r] + bias[1]);
            const float gt = ftanh   (acc[2][r] + bias[2]);
            const float os = fsigmoid(acc[3][r] + bias[3]);
            const float cn = fs * cc[r] + is * gt;
            cc[r] = cn;
            const float h = os * ftanh(cn);
            const int m = r + 8 * lhi;
            nxt[m * KTOT + FIN + col] = f2bf(h);
            if (t == TSEQ - 1) sH[m * HID + col] = h;
        }
    }

    __syncthreads();

    // ================= head: out[b, j] = h_last . W_fc[j] + b_fc[j] =================
    if (tid < 16 * LOUT) {
        const int m = tid / LOUT, j = tid % LOUT;
        float s = b_fc[j];
        const float* wf = W_fc + j * HID;
        #pragma unroll 8
        for (int n = 0; n < HID; ++n) s += sH[m * HID + n] * wf[n];
        out[(size_t)(row0 + m) * LOUT + j] = s;   // [B, L, 1] flat
    }
}

extern "C" void kernel_launch(void* const* d_in, const int* in_sizes, int n_in,
                              void* d_out, int out_size, void* d_ws, size_t ws_size,
                              hipStream_t stream) {
    const float* x    = (const float*)d_in[0];
    const float* W_ih = (const float*)d_in[1];
    const float* W_hh = (const float*)d_in[2];
    const float* b_ih = (const float*)d_in[3];
    const float* b_hh = (const float*)d_in[4];
    const float* W_fc = (const float*)d_in[5];
    const float* b_fc = (const float*)d_in[6];
    float* out = (float*)d_out;

    dim3 grid(BATCH / 16);   // 32 workgroups, one 16-row batch tile each
    dim3 block(256);         // 8 waves; wave w owns h columns [w*16, w*16+16)
    lstm_fused_wmma<<<grid, block, 0, stream>>>(x, W_ih, W_hh, b_ih, b_hh,
                                                W_fc, b_fc, out);
}
